// AdvancedGAT_LSTM_65884798320880
// MI455X (gfx1250) — compile-verified, matrix operand order checked by reading, not run on hardware
//
#include <hip/hip_runtime.h>

// ---------------------------------------------------------------------------
// Types for CDNA5 WMMA (gfx1250, wave32)
// ---------------------------------------------------------------------------
typedef __bf16 bf16_t;
typedef bf16_t v16bf __attribute__((ext_vector_type(16)));
typedef float  v8f   __attribute__((ext_vector_type(8)));

__device__ __forceinline__ unsigned short f2bf(float f) {
  unsigned u = __float_as_uint(f);
  unsigned r = (u + 0x7FFFu + ((u >> 16) & 1u)) >> 16;   // round-to-nearest-even
  return (unsigned short)r;
}

// ---------------------------------------------------------------------------
// bf16 WMMA GEMM: C[M,N] (f32) = A[M,K] (bf16, row-major) * B[K,N] (bf16, row-major)
// Requires K % 32 == 0.
// FULL=true : M % 128 == 0 && N % 128 == 0 (no guards, fully vectorized staging)
// FULL=false: arbitrary M,N (branchless clamp+select guards)
// Block: 256 threads (8 wave32), tile 128x128x32.
// Wave (wm in 0..1, wn in 0..3) computes a 64x32 sub-tile = 4x2 WMMA tiles.
// ---------------------------------------------------------------------------
#define BM 128
#define BN 128
#define BK 32

template <bool FULL>
__global__ __launch_bounds__(256) void gemm_bf16_wmma(
    const unsigned short* __restrict__ A,
    const unsigned short* __restrict__ B,
    float* __restrict__ C,
    int M, int N, int K)
{
  __shared__ unsigned short As[BM][BK + 8];   // [m][k]   row stride 80B (16B aligned)
  __shared__ unsigned short Bs[BN][BK + 8];   // transposed: [n][k]

  const int tid  = threadIdx.x;
  const int lane = tid & 31;
  const int wave = tid >> 5;
  const int wm   = wave >> 2;        // 0..1
  const int wn   = wave & 3;         // 0..3
  const int lm   = lane & 15;        // M/N index within 16x16 tile
  const int khi  = (lane >> 4) & 1;  // +8 to K for upper half-wave

  const int bm0 = blockIdx.y * BM;
  const int bn0 = blockIdx.x * BN;

  v8f acc[4][2];
#pragma unroll
  for (int i = 0; i < 4; ++i)
#pragma unroll
    for (int j = 0; j < 2; ++j)
#pragma unroll
      for (int r = 0; r < 8; ++r) acc[i][j][r] = 0.0f;

  for (int k0 = 0; k0 < K; k0 += BK) {
    if (FULL) {
      // ---- A tile: 128x32 bf16 = 512 uint4 units, 2 per thread
#pragma unroll
      for (int it = 0; it < 2; ++it) {
        int u = tid + it * 256;          // 0..511
        int r = u >> 2;                  // 0..127
        int k = (u & 3) << 3;            // 0,8,16,24
        uint4 v = *(const uint4*)(A + (size_t)(bm0 + r) * K + k0 + k);
        *(uint4*)&As[r][k] = v;
      }
      // ---- B tile: 32x128 bf16, one 4(k)x4(n) micro-block per thread,
      //      transposed in registers, stored as b64 to LDS
      {
        int kq = (tid >> 5) << 2;        // 0,4,...,28
        int nq = (tid & 31) << 2;        // 0,4,...,124
        unsigned short e[4][4];
#pragma unroll
        for (int dk = 0; dk < 4; ++dk) {
          uint2 v = *(const uint2*)(B + (size_t)(k0 + kq + dk) * N + bn0 + nq);
          e[dk][0] = (unsigned short)v.x;
          e[dk][1] = (unsigned short)(v.x >> 16);
          e[dk][2] = (unsigned short)v.y;
          e[dk][3] = (unsigned short)(v.y >> 16);
        }
#pragma unroll
        for (int dn = 0; dn < 4; ++dn) {
          uint2 o;
          o.x = (unsigned)e[0][dn] | ((unsigned)e[1][dn] << 16);
          o.y = (unsigned)e[2][dn] | ((unsigned)e[3][dn] << 16);
          *(uint2*)&Bs[nq + dn][kq] = o;
        }
      }
      // speculative prefetch of next A tile
      if (k0 + BK < K)
        __builtin_prefetch(A + (size_t)(bm0 + (tid >> 1)) * K + k0 + BK, 0, 0);
    } else {
      // ---- generic, branchless (clamp address, select zero)
#pragma unroll
      for (int it = 0; it < 8; ++it) {
        int u  = tid + it * 256;
        int r  = u >> 4;
        int kc = (u & 15) << 1;
        int gr = bm0 + r;
        size_t ga = (size_t)(gr < M ? gr : 0) * K + k0 + kc;
        unsigned val = *(const unsigned*)(A + ga);
        if (gr >= M) val = 0;
        *(unsigned*)&As[r][kc] = val;
      }
#pragma unroll
      for (int it = 0; it < 8; ++it) {
        int u  = tid + it * 256;
        int kr = u >> 6;                 // 0..31
        int nc = (u & 63) << 1;          // 0..126
        int gk = k0 + kr;
        int gn = bn0 + nc;
        int c0 = (gn < N) ? gn : (N - 1);
        int c1 = (gn + 1 < N) ? (gn + 1) : (N - 1);
        unsigned short e0 = B[(size_t)gk * N + c0];
        unsigned short e1 = B[(size_t)gk * N + c1];
        if (gn >= N)     e0 = 0;
        if (gn + 1 >= N) e1 = 0;
        Bs[nc][kr]     = e0;
        Bs[nc + 1][kr] = e1;
      }
    }
    __syncthreads();

    // ---- fragments (ISA 16-bit layout: lane=M/N, VGPR pair p -> K base below)
    union FragU { v16bf v; unsigned d[8]; };
    FragU afr[4], bfr[2];
#pragma unroll
    for (int i = 0; i < 4; ++i) {
      int r = wm * 64 + i * 16 + lm;
#pragma unroll
      for (int p = 0; p < 8; ++p) {
        int kk = ((p & 4) << 2) + (khi << 3) + ((p & 3) << 1);
        afr[i].d[p] = *(const unsigned*)&As[r][kk];
      }
    }
#pragma unroll
    for (int j = 0; j < 2; ++j) {
      int c = wn * 32 + j * 16 + lm;
#pragma unroll
      for (int p = 0; p < 8; ++p) {
        int kk = ((p & 4) << 2) + (khi << 3) + ((p & 3) << 1);
        bfr[j].d[p] = *(const unsigned*)&Bs[c][kk];
      }
    }
#pragma unroll
    for (int i = 0; i < 4; ++i)
#pragma unroll
      for (int j = 0; j < 2; ++j)
        acc[i][j] = __builtin_amdgcn_wmma_f32_16x16x32_bf16(
            false, afr[i].v, false, bfr[j].v, (short)0, acc[i][j], false, false);
    __syncthreads();
  }

  // ---- epilogue: C/D layout -> row = +khi*8 + vgpr r, col = lane&15
#pragma unroll
  for (int i = 0; i < 4; ++i) {
#pragma unroll
    for (int j = 0; j < 2; ++j) {
      int col = bn0 + wn * 32 + j * 16 + lm;
#pragma unroll
      for (int r = 0; r < 8; ++r) {
        int row = bm0 + wm * 64 + i * 16 + khi * 8 + r;
        if (FULL || (row < M && col < N)) C[(size_t)row * N + col] = acc[i][j][r];
      }
    }
  }
}

// ---------------------------------------------------------------------------
// Conversion / fill helpers
// ---------------------------------------------------------------------------
__global__ void padcols_bf16(const float* __restrict__ in, unsigned short* __restrict__ out,
                             int R, int K, int Kp) {
  long long idx = (long long)blockIdx.x * blockDim.x + threadIdx.x;
  long long total = (long long)R * Kp;
  if (idx >= total) return;
  int r = (int)(idx / Kp), c = (int)(idx % Kp);
  float v = (c < K) ? in[(size_t)r * K + c] : 0.0f;
  out[idx] = f2bf(v);
}

__global__ void padrows_bf16(const float* __restrict__ in, unsigned short* __restrict__ out,
                             int R, int Rp, int C) {
  long long idx = (long long)blockIdx.x * blockDim.x + threadIdx.x;
  long long total = (long long)Rp * C;
  if (idx >= total) return;
  int r = (int)(idx / C), c = (int)(idx % C);
  float v = (r < R) ? in[(size_t)r * C + c] : 0.0f;
  out[idx] = f2bf(v);
}

__global__ void transpose_bf16(const float* __restrict__ in, unsigned short* __restrict__ out,
                               int R, int C) {  // in [R,C] -> out [C,R]
  long long idx = (long long)blockIdx.x * blockDim.x + threadIdx.x;
  long long total = (long long)R * C;
  if (idx >= total) return;
  int r = (int)(idx / C), c = (int)(idx % C);
  out[(size_t)c * R + r] = f2bf(in[idx]);
}

__global__ void fill_f32(float* p, float v, long long n) {
  long long i = (long long)blockIdx.x * blockDim.x + threadIdx.x;
  if (i < n) p[i] = v;
}
__global__ void fill_u32(unsigned* p, unsigned v, long long n) {
  long long i = (long long)blockIdx.x * blockDim.x + threadIdx.x;
  if (i < n) p[i] = v;
}
__global__ void fill_u16(unsigned short* p, unsigned short v, long long n) {
  long long i = (long long)blockIdx.x * blockDim.x + threadIdx.x;
  if (i < n) p[i] = v;
}

// ---------------------------------------------------------------------------
// GAT attention kernels
// ---------------------------------------------------------------------------
__global__ void es_ed_kernel(const float* __restrict__ G, const float* __restrict__ as_,
                             const float* __restrict__ ad_, float* __restrict__ es,
                             float* __restrict__ ed, int N, int H, int D) {
  int idx = blockIdx.x * blockDim.x + threadIdx.x;
  if (idx >= N * H) return;
  int n = idx / H, h = idx % H;
  const float* gp = G + (size_t)n * H * D + (size_t)h * D;
  const float* ap = as_ + (size_t)h * D;
  const float* bp = ad_ + (size_t)h * D;
  float s = 0.f, t = 0.f;
  for (int d = 0; d < D; ++d) { float v = gp[d]; s += v * ap[d]; t += v * bp[d]; }
  es[idx] = s; ed[idx] = t;
}

__global__ void edge_logit_kernel(const int* __restrict__ src, const int* __restrict__ dst,
                                  const float* __restrict__ es, const float* __restrict__ ed,
                                  float* __restrict__ logit, unsigned* __restrict__ mkey,
                                  int E, int Etot, int H) {
  long long idx = (long long)blockIdx.x * blockDim.x + threadIdx.x;
  if (idx >= (long long)Etot * H) return;
  int e = (int)(idx / H), h = (int)(idx % H);
  int s_, d_;
  if (e < E) { s_ = src[e]; d_ = dst[e]; } else { s_ = d_ = e - E; }
  float l = es[s_ * H + h] + ed[d_ * H + h];
  l = (l > 0.f) ? l : 0.2f * l;                       // leaky_relu(0.2)
  logit[idx] = l;
  unsigned b = __float_as_uint(l);
  unsigned key = (b & 0x80000000u) ? ~b : (b | 0x80000000u);  // order-preserving
  atomicMax(&mkey[d_ * H + h], key);
}

__global__ void decode_max_kernel(const unsigned* __restrict__ mkey, float* __restrict__ mval, int n) {
  int i = blockIdx.x * blockDim.x + threadIdx.x;
  if (i >= n) return;
  unsigned k = mkey[i];
  unsigned b = (k & 0x80000000u) ? (k & 0x7FFFFFFFu) : ~k;
  mval[i] = __uint_as_float(b);
}

__global__ void edge_z_kernel(const int* __restrict__ dst, const float* __restrict__ logit,
                              const float* __restrict__ mval, float* __restrict__ z,
                              int E, int Etot, int H) {
  long long idx = (long long)blockIdx.x * blockDim.x + threadIdx.x;
  if (idx >= (long long)Etot * H) return;
  int e = (int)(idx / H), h = (int)(idx % H);
  int d_ = (e < E) ? dst[e] : (e - E);
  float p = __expf(logit[idx] - mval[d_ * H + h]);
  atomicAdd(&z[d_ * H + h], p);
}

__global__ void edge_agg_kernel(const int* __restrict__ src, const int* __restrict__ dst,
                                const float* __restrict__ logit, const float* __restrict__ mval,
                                const float* __restrict__ z, const float* __restrict__ G,
                                float* __restrict__ O, int E, int Etot, int H, int D, int concat) {
  long long idx = (long long)blockIdx.x * blockDim.x + threadIdx.x;
  long long total = (long long)Etot * H * D;
  if (idx >= total) return;
  int d = (int)(idx % D);
  long long eh = idx / D;
  int h = (int)(eh % H);
  int e = (int)(eh / H);
  int s_, dd;
  if (e < E) { s_ = src[e]; dd = dst[e]; } else { s_ = dd = e - E; }
  float alpha = __expf(logit[(size_t)e * H + h] - mval[dd * H + h]) / z[dd * H + h];
  float v = alpha * G[(size_t)s_ * H * D + (size_t)h * D + d];
  if (concat) atomicAdd(&O[(size_t)dd * H * D + (size_t)h * D + d], v);
  else        atomicAdd(&O[(size_t)dd * D + d], v * (1.0f / H));
}

// ELU(x + bias) -> LayerNorm(g, be) -> (+ residual). One block per node.
__global__ void post_gat_kernel(const float* __restrict__ O, const float* __restrict__ bias,
                                const float* __restrict__ g, const float* __restrict__ be,
                                const float* __restrict__ res, float* __restrict__ out, int D) {
  __shared__ float s1[256], s2[256];
  int n = blockIdx.x, tid = threadIdx.x;
  const float* op = O + (size_t)n * D;
  float* outp = out + (size_t)n * D;
  float lsum = 0.f, lsq = 0.f;
  for (int d = tid; d < D; d += 256) {
    float v = op[d] + bias[d];
    v = (v > 0.f) ? v : (__expf(v) - 1.0f);   // ELU
    outp[d] = v;
    lsum += v; lsq += v * v;
  }
  s1[tid] = lsum; s2[tid] = lsq;
  __syncthreads();
  for (int s = 128; s > 0; s >>= 1) {
    if (tid < s) { s1[tid] += s1[tid + s]; s2[tid] += s2[tid + s]; }
    __syncthreads();
  }
  float mu  = s1[0] / D;
  float var = s2[0] / D - mu * mu;
  float inv = rsqrtf(var + 1e-5f);
  __syncthreads();
  for (int d = tid; d < D; d += 256) {
    float y = (outp[d] - mu) * inv * g[d] + be[d];
    if (res) y += res[(size_t)n * D + d];
    outp[d] = y;
  }
}

// ---------------------------------------------------------------------------
// LSTM kernels
// ---------------------------------------------------------------------------
// xs[(t*B+b), d] = x4[(b*T+t), d]   (bf16)
__global__ void x4_to_seq_bf16(const float* __restrict__ x4, unsigned short* __restrict__ out,
                               int T, int B, int D) {
  long long idx = (long long)blockIdx.x * blockDim.x + threadIdx.x;
  long long total = (long long)T * B * D;
  if (idx >= total) return;
  int d = (int)(idx % D);
  int row = (int)(idx / D);
  int t = row / B, b = row % B;
  out[idx] = f2bf(x4[((size_t)b * T + t) * D + d]);
}

__global__ void lstm_gates_kernel(const float* __restrict__ XWt, const float* __restrict__ HW,
                                  const float* __restrict__ bih, const float* __restrict__ bhh,
                                  float* __restrict__ c, float* __restrict__ h,
                                  unsigned short* __restrict__ hbf, int B, int Hh) {
  int idx = blockIdx.x * blockDim.x + threadIdx.x;
  if (idx >= B * Hh) return;
  int b = idx / Hh, j = idx % Hh;
  const float* xr = XWt + (size_t)b * 4 * Hh;
  const float* hr = HW + (size_t)b * 4 * Hh;
  float gi = xr[j]          + hr[j]          + bih[j]          + bhh[j];
  float gf = xr[Hh + j]     + hr[Hh + j]     + bih[Hh + j]     + bhh[Hh + j];
  float gg = xr[2 * Hh + j] + hr[2 * Hh + j] + bih[2 * Hh + j] + bhh[2 * Hh + j];
  float go = xr[3 * Hh + j] + hr[3 * Hh + j] + bih[3 * Hh + j] + bhh[3 * Hh + j];
  float si = 1.f / (1.f + __expf(-gi));
  float sf = 1.f / (1.f + __expf(-gf));
  float so = 1.f / (1.f + __expf(-go));
  float tg = tanhf(gg);
  float cn = sf * c[idx] + si * tg;
  float hn = so * tanhf(cn);
  c[idx] = cn; h[idx] = hn; hbf[idx] = f2bf(hn);
}

__global__ void concat_bf16_kernel(const float* __restrict__ hF, const float* __restrict__ hB,
                                   unsigned short* __restrict__ out, int B, int Hh) {
  int idx = blockIdx.x * blockDim.x + threadIdx.x;
  if (idx >= B * 2 * Hh) return;
  int b = idx / (2 * Hh), j = idx % (2 * Hh);
  float v = (j < Hh) ? hF[(size_t)b * Hh + j] : hB[(size_t)b * Hh + (j - Hh)];
  out[idx] = f2bf(v);
}

__global__ void bias_relu_bf16_kernel(const float* __restrict__ in, const float* __restrict__ bias,
                                      unsigned short* __restrict__ out, int R, int D) {
  int idx = blockIdx.x * blockDim.x + threadIdx.x;
  if (idx >= R * D) return;
  int d = idx % D;
  float v = in[idx] + bias[d];
  out[idx] = f2bf(v > 0.f ? v : 0.f);
}

__global__ void bias_add_out_kernel(const float* __restrict__ in, const float* __restrict__ bias,
                                    float* __restrict__ out, int R, int D) {
  int idx = blockIdx.x * blockDim.x + threadIdx.x;
  if (idx >= R * D) return;
  int d = idx % D;
  out[idx] = in[idx] + bias[d];
}

// ---------------------------------------------------------------------------
// Host orchestration
// ---------------------------------------------------------------------------
static inline unsigned nb(long long n) { return (unsigned)((n + 255) / 256); }

static void launch_gemm(const unsigned short* A, const unsigned short* B, float* C,
                        int M, int N, int K, hipStream_t s) {
  dim3 grid((N + BN - 1) / BN, (M + BM - 1) / BM);
  if ((M % BM) == 0 && (N % BN) == 0)
    gemm_bf16_wmma<true><<<grid, 256, 0, s>>>(A, B, C, M, N, K);
  else
    gemm_bf16_wmma<false><<<grid, 256, 0, s>>>(A, B, C, M, N, K);
}

extern "C" void kernel_launch(void* const* d_in, const int* in_sizes, int n_in,
                              void* d_out, int out_size, void* d_ws, size_t ws_size,
                              hipStream_t stream) {
  (void)n_in; (void)out_size; (void)ws_size;
  const float* x   = (const float*)d_in[0];
  const int*   src = (const int*)d_in[1];
  const int*   dst = (const int*)d_in[2];
  const float* W1  = (const float*)d_in[3];
  const float* as1 = (const float*)d_in[4];  const float* ad1 = (const float*)d_in[5];
  const float* b1  = (const float*)d_in[6];  const float* g1  = (const float*)d_in[7];
  const float* be1 = (const float*)d_in[8];
  const float* W2  = (const float*)d_in[9];
  const float* as2 = (const float*)d_in[10]; const float* ad2 = (const float*)d_in[11];
  const float* b2  = (const float*)d_in[12]; const float* g2  = (const float*)d_in[13];
  const float* be2 = (const float*)d_in[14];
  const float* W3  = (const float*)d_in[15];
  const float* as3 = (const float*)d_in[16]; const float* ad3 = (const float*)d_in[17];
  const float* b3  = (const float*)d_in[18]; const float* g3  = (const float*)d_in[19];
  const float* be3 = (const float*)d_in[20];
  const float* W4  = (const float*)d_in[21];
  const float* as4 = (const float*)d_in[22]; const float* ad4 = (const float*)d_in[23];
  const float* b4  = (const float*)d_in[24]; const float* g4  = (const float*)d_in[25];
  const float* be4 = (const float*)d_in[26];
  const float* WihF = (const float*)d_in[27]; const float* WhhF = (const float*)d_in[28];
  const float* bihF = (const float*)d_in[29]; const float* bhhF = (const float*)d_in[30];
  const float* WihB = (const float*)d_in[31]; const float* WhhB = (const float*)d_in[32];
  const float* bihB = (const float*)d_in[33]; const float* bhhB = (const float*)d_in[34];
  const float* fc1w = (const float*)d_in[35]; const float* fc1b = (const float*)d_in[36];
  const float* fc2w = (const float*)d_in[37]; const float* fc2b = (const float*)d_in[38];

  const int N = 8192, H = 8, T = 128, B = 64;
  const int E = in_sizes[1];          // 49152
  const int Etot = E + N;

  // ---- workspace bump allocator
  char* wsp = (char*)d_ws;
  size_t off = 0;
  auto alloc = [&](size_t bytes) -> void* {
    void* p = wsp + off;
    off += (bytes + 255) & ~((size_t)255);
    return p;
  };
  unsigned short* wW1   = (unsigned short*)alloc(320ull * 2048 * 2);
  unsigned short* wW2   = (unsigned short*)alloc(2048ull * 2048 * 2);
  unsigned short* wW3   = (unsigned short*)alloc(2048ull * 2048 * 2);
  unsigned short* wW4   = (unsigned short*)alloc(2048ull * 8192 * 2);
  unsigned short* wWihF = (unsigned short*)alloc(1024ull * 4096 * 2);
  unsigned short* wWhhF = (unsigned short*)alloc(1024ull * 4096 * 2);
  unsigned short* wWihB = (unsigned short*)alloc(1024ull * 4096 * 2);
  unsigned short* wWhhB = (unsigned short*)alloc(1024ull * 4096 * 2);
  unsigned short* wFc1  = (unsigned short*)alloc(2048ull * 512 * 2);
  unsigned short* wFc2  = (unsigned short*)alloc(512ull * 104 * 2);
  unsigned short* aBuf  = (unsigned short*)alloc(8192ull * 2048 * 2);
  float* G    = (float*)alloc(8192ull * 8192 * 4);
  float* O    = (float*)alloc(8192ull * 2048 * 4);
  float* x1   = (float*)alloc(8192ull * 2048 * 4);
  float* x2   = (float*)alloc(8192ull * 2048 * 4);
  float* x3   = (float*)alloc(8192ull * 2048 * 4);
  float* x4v  = (float*)alloc(8192ull * 1024 * 4);
  float* esb  = (float*)alloc((size_t)N * H * 4);
  float* edb  = (float*)alloc((size_t)N * H * 4);
  float* mval = (float*)alloc((size_t)N * H * 4);
  float* zb   = (float*)alloc((size_t)N * H * 4);
  unsigned* mkey = (unsigned*)alloc((size_t)N * H * 4);
  float* logit = (float*)alloc((size_t)Etot * H * 4);
  float* XW   = (float*)alloc(8192ull * 4096 * 4);
  float* HWb  = (float*)alloc(64ull * 4096 * 4);
  float* hFv  = (float*)alloc(64ull * 1024 * 4);
  float* cFv  = (float*)alloc(64ull * 1024 * 4);
  float* hBv  = (float*)alloc(64ull * 1024 * 4);
  float* cBv  = (float*)alloc(64ull * 1024 * 4);
  unsigned short* hbf  = (unsigned short*)alloc(64ull * 1024 * 2);
  unsigned short* catb = (unsigned short*)alloc(64ull * 2048 * 2);
  float* f1 = (float*)alloc(64ull * 512 * 4);
  unsigned short* f1b = (unsigned short*)alloc(64ull * 512 * 2);
  float* f2 = (float*)alloc(64ull * 104 * 4);

  // ---- convert weights to bf16 (pad/transpose as needed)
  padrows_bf16<<<nb(320ll * 2048), 256, 0, stream>>>(W1, wW1, 300, 320, 2048);
  padrows_bf16<<<nb(2048ll * 2048), 256, 0, stream>>>(W2, wW2, 2048, 2048, 2048);
  padrows_bf16<<<nb(2048ll * 2048), 256, 0, stream>>>(W3, wW3, 2048, 2048, 2048);
  padrows_bf16<<<nb(2048ll * 8192), 256, 0, stream>>>(W4, wW4, 2048, 2048, 8192);
  transpose_bf16<<<nb(4096ll * 1024), 256, 0, stream>>>(WihF, wWihF, 4096, 1024);
  transpose_bf16<<<nb(4096ll * 1024), 256, 0, stream>>>(WhhF, wWhhF, 4096, 1024);
  transpose_bf16<<<nb(4096ll * 1024), 256, 0, stream>>>(WihB, wWihB, 4096, 1024);
  transpose_bf16<<<nb(4096ll * 1024), 256, 0, stream>>>(WhhB, wWhhB, 4096, 1024);
  padrows_bf16<<<nb(2048ll * 512), 256, 0, stream>>>(fc1w, wFc1, 2048, 2048, 512);
  padrows_bf16<<<nb(512ll * 104), 256, 0, stream>>>(fc2w, wFc2, 512, 512, 104);

  // ---- one GAT layer
  auto gat_layer = [&](const float* xin, int K, int Kp, const unsigned short* Wbf,
                       const float* as_, const float* ad_, const float* bias,
                       const float* g_, const float* be_, const float* res,
                       float* xout, int D, int concat) {
    const int HD = H * D;
    padcols_bf16<<<nb((long long)N * Kp), 256, 0, stream>>>(xin, aBuf, N, K, Kp);
    launch_gemm(aBuf, Wbf, G, N, HD, Kp, stream);                       // G = h features
    es_ed_kernel<<<nb((long long)N * H), 256, 0, stream>>>(G, as_, ad_, esb, edb, N, H, D);
    fill_u32<<<nb((long long)N * H), 256, 0, stream>>>(mkey, 0x007FFFFFu, (long long)N * H); // key(-inf)
    fill_f32<<<nb((long long)N * H), 256, 0, stream>>>(zb, 0.f, (long long)N * H);
    edge_logit_kernel<<<nb((long long)Etot * H), 256, 0, stream>>>(src, dst, esb, edb, logit, mkey, E, Etot, H);
    decode_max_kernel<<<nb((long long)N * H), 256, 0, stream>>>(mkey, mval, N * H);
    edge_z_kernel<<<nb((long long)Etot * H), 256, 0, stream>>>(dst, logit, mval, zb, E, Etot, H);
    long long osz = concat ? (long long)N * HD : (long long)N * D;
    fill_f32<<<nb(osz), 256, 0, stream>>>(O, 0.f, osz);
    edge_agg_kernel<<<nb((long long)Etot * HD), 256, 0, stream>>>(src, dst, logit, mval, zb, G, O,
                                                                  E, Etot, H, D, concat);
    post_gat_kernel<<<N, 256, 0, stream>>>(O, bias, g_, be_, res, xout, concat ? HD : D);
  };

  gat_layer(x,   300, 320,  wW1, as1, ad1, b1, g1, be1, nullptr, x1, 256, 1);
  gat_layer(x1, 2048, 2048, wW2, as2, ad2, b2, g2, be2, x1,      x2, 256, 1);
  gat_layer(x2, 2048, 2048, wW3, as3, ad3, b3, g3, be3, x2,      x3, 256, 1);
  gat_layer(x3, 2048, 2048, wW4, as4, ad4, b4, g4, be4, nullptr, x4v, 1024, 0);

  // ---- BiLSTM over [T=128, B=64, 1024]
  x4_to_seq_bf16<<<nb(8192ll * 1024), 256, 0, stream>>>(x4v, aBuf, T, B, 1024);

  // forward direction
  launch_gemm(aBuf, wWihF, XW, T * B, 4096, 1024, stream);     // XW[t*B+b, 4096]
  fill_f32<<<nb(64ll * 1024), 256, 0, stream>>>(hFv, 0.f, 64ll * 1024);
  fill_f32<<<nb(64ll * 1024), 256, 0, stream>>>(cFv, 0.f, 64ll * 1024);
  fill_u16<<<nb(64ll * 1024), 256, 0, stream>>>(hbf, 0, 64ll * 1024);
  for (int t = 0; t < T; ++t) {
    launch_gemm(hbf, wWhhF, HWb, B, 4096, 1024, stream);
    lstm_gates_kernel<<<nb(64ll * 1024), 256, 0, stream>>>(XW + (size_t)t * B * 4096, HWb,
                                                           bihF, bhhF, cFv, hFv, hbf, B, 1024);
  }
  // backward direction (reuse XW with Wih_b; step s consumes time t = T-1-s)
  launch_gemm(aBuf, wWihB, XW, T * B, 4096, 1024, stream);
  fill_f32<<<nb(64ll * 1024), 256, 0, stream>>>(hBv, 0.f, 64ll * 1024);
  fill_f32<<<nb(64ll * 1024), 256, 0, stream>>>(cBv, 0.f, 64ll * 1024);
  fill_u16<<<nb(64ll * 1024), 256, 0, stream>>>(hbf, 0, 64ll * 1024);
  for (int s = 0; s < T; ++s) {
    int t = T - 1 - s;
    launch_gemm(hbf, wWhhB, HWb, B, 4096, 1024, stream);
    lstm_gates_kernel<<<nb(64ll * 1024), 256, 0, stream>>>(XW + (size_t)t * B * 4096, HWb,
                                                           bihB, bhhB, cBv, hBv, hbf, B, 1024);
  }

  // ---- head: concat -> fc1 + relu -> fc2 + bias -> d_out
  concat_bf16_kernel<<<nb(64ll * 2048), 256, 0, stream>>>(hFv, hBv, catb, B, 1024);
  launch_gemm(catb, wFc1, f1, B, 512, 2048, stream);
  bias_relu_bf16_kernel<<<nb(64ll * 512), 256, 0, stream>>>(f1, fc1b, f1b, B, 512);
  launch_gemm(f1b, wFc2, f2, B, 104, 512, stream);
  bias_add_out_kernel<<<nb(64ll * 104), 256, 0, stream>>>(f2, fc2b, (float*)d_out, B, 104);
}
